// ResABlock_58437325029409
// MI455X (gfx1250) — matrix operand order repaired and checked by reading, not compiled, not run on hardware
//
#include <hip/hip_runtime.h>
#include <hip/hip_bf16.h>
#include <math.h>

// ---------------------------------------------------------------------------
// Types for CDNA5 WMMA (gfx1250, wave32)
// ---------------------------------------------------------------------------
typedef __bf16 bf16x16 __attribute__((ext_vector_type(16)));
typedef __bf16 bf16x8  __attribute__((ext_vector_type(8)));
typedef float  f32x8   __attribute__((ext_vector_type(8)));
typedef float  f32x4   __attribute__((ext_vector_type(4)));

static __device__ __forceinline__ f32x8 wmma_bf16(bf16x16 a, bf16x16 b, f32x8 c) {
  // D = A(16x32 bf16) * B(32x16 bf16) + C(16x16 f32)
  return __builtin_amdgcn_wmma_f32_16x16x32_bf16(
      /*neg_a=*/false, a, /*neg_b=*/false, b,
      /*c_mod=*/(short)0, c, /*reuse_a=*/false, /*reuse_b=*/false);
}

// gfx1250 async DMA: global -> LDS, 16B per lane, tracked by ASYNCcnt.
// LDS destination address = low 32 bits of the generic pointer (ISA 10.2:
// LDS aperture addresses truncate to the LDS byte offset).
static __device__ __forceinline__ void async_copy_b128(unsigned lds_off, const void* gptr) {
  asm volatile("global_load_async_to_lds_b128 %0, %1, off"
               :: "v"(lds_off), "v"((unsigned long long)(uintptr_t)gptr)
               : "memory");
}
static __device__ __forceinline__ void wait_async0() {
  asm volatile("s_wait_asynccnt 0x0" ::: "memory");
}

// ---------------------------------------------------------------------------
// Problem constants
// ---------------------------------------------------------------------------
#define BATCH 2
#define SEQ   2048
#define DMODEL 512
#define NHEAD 8
#define DHEAD 64
#define HIDDEN 2048
#define MROWS (BATCH * SEQ)   // 4096
#define LN_EPS 1e-5f

// ---------------------------------------------------------------------------
// Precision plan (roofline): every GEMM/attention operand is stored ONCE in
// bf16 by its producer (weights pre-transposed [N][K] by a one-shot kernel,
// activations by the previous stage's epilogue). This halves the HBM bytes of
// every matrix input on a 23.3 TB/s part where these stages are memory-bound,
// and makes LDS staging a pure byte copy -> async DMA engine, no VGPR bounce.
// ---------------------------------------------------------------------------

// ---------------------------------------------------------------------------
// GEMM: C[M,N] = epilogue(A[M,K] @ Wt[N,K]^T + bias[N] [+ residual])
// A bf16 [M,K], Wt bf16 [N,K] (pre-transposed). Double-buffered LDS staged
// with global_load_async_to_lds_b128. 256 threads = 8 waves, 4(M)x2(N);
// each wave: 32x32 tile = 2x2 WMMA per 32-k step, 8 WMMA per 64-k stage.
// ---------------------------------------------------------------------------
#define BM 128
#define BN 64
#define BK 64
#define LPAD 8   // row stride 72 elems = 144B: 16B aligned, conflict-reducing
#define LROW (BK + LPAD)

__global__ __launch_bounds__(256)
void gemm_bf16_wmma(const __bf16* __restrict__ A,    // [M,K] bf16
                    const __bf16* __restrict__ Wt,   // [N,K] bf16 (transposed)
                    const float* __restrict__ bias,
                    const float* __restrict__ residual,  // f32 [M,N] or nullptr
                    void* __restrict__ Cout,             // f32* or bf16*
                    int M, int N, int K, int apply_gelu, int out_bf16)
{
  __shared__ __bf16 As[2][BM][LROW];  // (36 KB)
  __shared__ __bf16 Ws[2][BN][LROW];  // (18 KB)

  const int tid     = threadIdx.x;
  const int lane    = tid & 31;
  const int wave    = tid >> 5;
  const int waveM   = wave >> 1;     // 0..3
  const int waveN   = wave & 1;      // 0..1
  const int l15     = lane & 15;
  const int halfSel = lane >> 4;     // 0 (lanes 0-15) or 1 (lanes 16-31)
  const int blockRow = blockIdx.y * BM;
  const int blockCol = blockIdx.x * BN;

  // staging geometry: 16B segment id = tid + i*256; row = id>>3, seg = id&7
  const int sRow = tid >> 3;          // 0..31 (A: +32*i, W: +32*i)
  const int sSeg = (tid & 7) * 8;     // bf16 elems (16B granules)

  f32x8 acc[2][2] = {};

  // ---- async-DMA one K-tile into LDS buffer `buf` (6 b128 per thread) ----
  auto issueAsync = [&](int buf, int k0) {
    const __bf16* ag = A + (size_t)(blockRow + sRow) * K + k0 + sSeg;
    unsigned al = (unsigned)(uintptr_t)&As[buf][sRow][sSeg];
    #pragma unroll
    for (int i = 0; i < 4; ++i)
      async_copy_b128(al + i * 32 * LROW * 2, ag + (size_t)i * 32 * K);
    const __bf16* wg = Wt + (size_t)(blockCol + sRow) * K + k0 + sSeg;
    unsigned wl = (unsigned)(uintptr_t)&Ws[buf][sRow][sSeg];
    #pragma unroll
    for (int i = 0; i < 2; ++i)
      async_copy_b128(wl + i * 32 * LROW * 2, wg + (size_t)i * 32 * K);
  };

  // ---- 8 WMMAs off one LDS buffer (2 k-steps of 32) ----
  auto compute = [&](int buf) {
    #pragma unroll
    for (int ks = 0; ks < 2; ++ks) {
      bf16x16 afr[2], bfr[2];
      #pragma unroll
      for (int i = 0; i < 2; ++i) {
        // A frag: lane row=l15; k = halfSel*16 + 0..15 contiguous
        const __bf16* ap = &As[buf][waveM * 32 + i * 16 + l15][ks * 32 + halfSel * 16];
        bf16x8 lo = *(const bf16x8*)ap;
        bf16x8 hi = *(const bf16x8*)(ap + 8);
        afr[i] = __builtin_shufflevector(lo, hi, 0,1,2,3,4,5,6,7,8,9,10,11,12,13,14,15);
      }
      #pragma unroll
      for (int j = 0; j < 2; ++j) {
        // B frag: lane col=l15; k = halfSel*16 + 0..15 contiguous in Ws[col][]
        const __bf16* bp = &Ws[buf][waveN * 32 + j * 16 + l15][ks * 32 + halfSel * 16];
        bf16x8 lo = *(const bf16x8*)bp;
        bf16x8 hi = *(const bf16x8*)(bp + 8);
        bfr[j] = __builtin_shufflevector(lo, hi, 0,1,2,3,4,5,6,7,8,9,10,11,12,13,14,15);
      }
      #pragma unroll
      for (int i = 0; i < 2; ++i)
        #pragma unroll
        for (int j = 0; j < 2; ++j)
          acc[i][j] = wmma_bf16(afr[i], bfr[j], acc[i][j]);
    }
  };

  // ---- software pipeline: DMA tile kt+1 while WMMAs chew tile kt ----
  issueAsync(0, 0);
  wait_async0();
  __syncthreads();

  const int nTiles = K / BK;
  for (int kt = 0; kt < nTiles; ++kt) {
    const int cur = kt & 1;
    const bool hasNext = (kt + 1 < nTiles);
    if (hasNext) issueAsync(cur ^ 1, (kt + 1) * BK);   // DMA in flight...
    if (kt + 2 < nTiles)                               // warm L2 a tile ahead
      __builtin_prefetch(A + (size_t)(blockRow + sRow) * K + (kt + 2) * BK + sSeg, 0, 1);
    compute(cur);                                      // ...during WMMAs
    if (hasNext) {
      wait_async0();
      __syncthreads();
    }
  }

  // ---- epilogue: bias (+residual f32) (+exact GELU), write f32 or bf16 ----
  #pragma unroll
  for (int i = 0; i < 2; ++i) {
    #pragma unroll
    for (int j = 0; j < 2; ++j) {
      int c = blockCol + waveN * 32 + j * 16 + l15;
      float bc = bias[c];
      #pragma unroll
      for (int v = 0; v < 8; ++v) {
        int r = blockRow + waveM * 32 + i * 16 + v + halfSel * 8;
        float val = acc[i][j][v] + bc;
        if (residual) val += residual[(size_t)r * N + c];
        if (apply_gelu) val = 0.5f * val * (1.0f + erff(val * 0.70710678118654752f));
        if (out_bf16) ((__bf16*)Cout)[(size_t)r * N + c] = (__bf16)val;
        else          ((float*)Cout)[(size_t)r * N + c] = val;
      }
    }
  }
}

// ---------------------------------------------------------------------------
// Flash attention, one wave (32 threads) per (b, h, 16 query rows).
//   energy = scale * Q K^T + prev   (written out: required output)
//   attn   = softmax(energy)        (online, in-register)
//   ctx    = attn @ V               (WMMA, fp32 accumulate, bf16 store)
// kqv is bf16 -> fragment loads are direct 16B vector loads, no conversion.
// K/V re-reads live in the 192 MB L2 (8 MB working set), so HBM traffic
// ~= prev read (268 MB) + energy write (268 MB), the mandatory minimum.
// ---------------------------------------------------------------------------
__global__ __launch_bounds__(32)
void attn_flash(const __bf16* __restrict__ kqv,  // [B,S,3D] bf16, order k|q|v
                const float* __restrict__ prev,  // [B,H,S,S]
                float* __restrict__ energy,      // [B,H,S,S] output
                __bf16* __restrict__ ctx)        // [B,S,D] bf16
{
  const int qt  = blockIdx.x & (SEQ / 16 - 1);
  const int bh  = blockIdx.x / (SEQ / 16);
  const int h   = bh & (NHEAD - 1);
  const int b   = bh / NHEAD;
  const int q0  = qt * 16;
  const int lane = threadIdx.x;
  const int l15 = lane & 15;
  const int halfSel = lane >> 4;
  const int threeD = 3 * DMODEL;
  const float scale = 0.125f;   // 1/sqrt(64)

  const __bf16* kbase = kqv + (size_t)b * SEQ * threeD + h * DHEAD;
  const __bf16* qbase = kqv + (size_t)b * SEQ * threeD + DMODEL + h * DHEAD;
  const __bf16* vbase = kqv + (size_t)b * SEQ * threeD + 2 * DMODEL + h * DHEAD;
  const float* prevb = prev   + (((size_t)b * NHEAD + h) * SEQ + q0) * SEQ;
  float*       eout  = energy + (((size_t)b * NHEAD + h) * SEQ + q0) * SEQ;

  __shared__ __bf16 Plds[16][32];   // probability tile, restaged for A-frag layout

  // Q as two A-fragments (d-chunks of 32); lane row = q0 + l15
  bf16x16 qa[2];
  {
    const __bf16* qrow = qbase + (size_t)(q0 + l15) * threeD;
    #pragma unroll
    for (int c = 0; c < 2; ++c) {
      const __bf16* qp = qrow + c * 32 + halfSel * 16;
      bf16x8 lo = *(const bf16x8*)qp;
      bf16x8 hi = *(const bf16x8*)(qp + 8);
      qa[c] = __builtin_shufflevector(lo, hi, 0,1,2,3,4,5,6,7,8,9,10,11,12,13,14,15);
    }
  }

  float m_run[8], l_run[8];
  #pragma unroll
  for (int v = 0; v < 8; ++v) { m_run[v] = -1e30f; l_run[v] = 0.0f; }
  f32x8 acc[4] = {};   // ctx accumulators: 4 d-tiles of 16

  for (int s0 = 0; s0 < SEQ; s0 += 32) {
    // ---- energy tile: 16 q-rows x 32 k-cols, 2 C tiles; K frags direct ----
    f32x8 e[2] = {};
    #pragma unroll
    for (int nt = 0; nt < 2; ++nt) {
      const int sk = s0 + nt * 16 + l15;              // B-frag: lane holds col sk
      #pragma unroll
      for (int c = 0; c < 2; ++c) {
        const __bf16* kp = kbase + (size_t)sk * threeD + c * 32 + halfSel * 16;
        bf16x8 lo = *(const bf16x8*)kp;
        bf16x8 hi = *(const bf16x8*)(kp + 8);
        bf16x16 kf = __builtin_shufflevector(lo, hi, 0,1,2,3,4,5,6,7,8,9,10,11,12,13,14,15);
        e[nt] = wmma_bf16(qa[c], kf, e[nt]);
      }
    }

    // ---- scale + prev, write energy, online softmax ----
    float pv[2][8];
    #pragma unroll
    for (int v = 0; v < 8; ++v) {
      const int r = v + halfSel * 8;                  // q row within tile
      const size_t ro = (size_t)r * SEQ + s0 + l15;
      float e0 = e[0][v] * scale + prevb[ro];
      float e1 = e[1][v] * scale + prevb[ro + 16];
      eout[ro]      = e0;
      eout[ro + 16] = e1;
      float mx = fmaxf(e0, e1);
      #pragma unroll
      for (int off = 1; off < 16; off <<= 1) mx = fmaxf(mx, __shfl_xor(mx, off, 32));
      float nm = fmaxf(m_run[v], mx);
      float p0 = __expf(e0 - nm);
      float p1 = __expf(e1 - nm);
      pv[0][v] = p0; pv[1][v] = p1;
      float rs = p0 + p1;
      #pragma unroll
      for (int off = 1; off < 16; off <<= 1) rs += __shfl_xor(rs, off, 32);
      float alpha = __expf(m_run[v] - nm);
      l_run[v] = l_run[v] * alpha + rs;
      m_run[v] = nm;
      #pragma unroll
      for (int t = 0; t < 4; ++t) acc[t][v] *= alpha;
    }

    // ---- restage P through LDS into A-fragment layout ----
    __syncthreads();
    #pragma unroll
    for (int nt = 0; nt < 2; ++nt)
      #pragma unroll
      for (int v = 0; v < 8; ++v)
        Plds[v + halfSel * 8][nt * 16 + l15] = (__bf16)pv[nt][v];
    __syncthreads();

    bf16x16 pa;
    {
      const __bf16* pp = &Plds[l15][halfSel * 16];
      bf16x8 lo = *(const bf16x8*)pp;
      bf16x8 hi = *(const bf16x8*)(pp + 8);
      pa = __builtin_shufflevector(lo, hi, 0,1,2,3,4,5,6,7,8,9,10,11,12,13,14,15);
    }

    // ---- ctx += P (16x32) @ V (32x64): 4 WMMAs ----
    #pragma unroll
    for (int t = 0; t < 4; ++t) {
      bf16x16 vf;
      const int d = t * 16 + l15;                     // B-frag: lane holds col d
      const __bf16* vp = vbase + (size_t)(s0 + halfSel * 16) * threeD + d;
      #pragma unroll
      for (int eIdx = 0; eIdx < 16; ++eIdx)
        vf[eIdx] = vp[(size_t)eIdx * threeD];
      acc[t] = wmma_bf16(pa, vf, acc[t]);
    }
    __syncthreads();
  }

  // ---- normalize and write ctx [B,S,D] bf16, layout h*DH + d ----
  #pragma unroll
  for (int t = 0; t < 4; ++t) {
    #pragma unroll
    for (int v = 0; v < 8; ++v) {
      const int r = v + halfSel * 8;
      ctx[((size_t)b * SEQ + q0 + r) * DMODEL + h * DHEAD + t * 16 + l15] =
          (__bf16)(acc[t][v] / l_run[v]);
    }
  }
}

// ---------------------------------------------------------------------------
// LayerNorm over last dim (D=512); optional secondary bf16 output for the
// next GEMM's A operand (halves its load traffic).
// ---------------------------------------------------------------------------
__global__ __launch_bounds__(256)
void ln_kernel(const float* __restrict__ in, const float* __restrict__ g,
               const float* __restrict__ bta, float* __restrict__ out,
               __bf16* __restrict__ out_bf, int Dm)
{
  __shared__ float shs[8], shs2[8];
  const int row = blockIdx.x;
  const float* xp = in + (size_t)row * Dm;
  float* op = out + (size_t)row * Dm;

  float s = 0.0f, s2 = 0.0f;
  for (int i = threadIdx.x; i < Dm; i += 256) { float v = xp[i]; s += v; s2 += v * v; }
  #pragma unroll
  for (int off = 16; off > 0; off >>= 1) {
    s  += __shfl_xor(s,  off, 32);
    s2 += __shfl_xor(s2, off, 32);
  }
  const int wv = threadIdx.x >> 5, ln = threadIdx.x & 31;
  if (ln == 0) { shs[wv] = s; shs2[wv] = s2; }
  __syncthreads();
  if (threadIdx.x < 32) {
    s  = (ln < 8) ? shs[ln]  : 0.0f;
    s2 = (ln < 8) ? shs2[ln] : 0.0f;
    #pragma unroll
    for (int off = 16; off > 0; off >>= 1) {
      s  += __shfl_xor(s,  off, 32);
      s2 += __shfl_xor(s2, off, 32);
    }
    if (ln == 0) { shs[0] = s; shs2[0] = s2; }
  }
  __syncthreads();
  const float mean = shs[0] / (float)Dm;
  const float var  = shs2[0] / (float)Dm - mean * mean;
  const float rstd = rsqrtf(var + LN_EPS);
  for (int i = threadIdx.x; i < Dm; i += 256) {
    float val = (xp[i] - mean) * rstd * g[i] + bta[i];
    op[i] = val;
    if (out_bf) out_bf[(size_t)row * Dm + i] = (__bf16)val;
  }
}

// ---------------------------------------------------------------------------
// One-shot converters: f32 -> bf16 (flat) and f32 [K,N] -> bf16 [N,K].
// ---------------------------------------------------------------------------
__global__ __launch_bounds__(256)
void cvt_bf16_kernel(const float* __restrict__ in, __bf16* __restrict__ out, int n4)
{
  int i = blockIdx.x * 256 + threadIdx.x;
  if (i < n4) {
    f32x4 v = ((const f32x4*)in)[i];
    __bf16* o = out + (size_t)i * 4;
    o[0] = (__bf16)v[0]; o[1] = (__bf16)v[1];
    o[2] = (__bf16)v[2]; o[3] = (__bf16)v[3];
  }
}

__global__ __launch_bounds__(256)
void transpose_cvt_kernel(const float* __restrict__ W, __bf16* __restrict__ Wt,
                          int K, int N)  // W[K,N] -> Wt[N,K]
{
  __shared__ float tile[32][33];
  const int nb = blockIdx.x * 32, kb = blockIdx.y * 32;
  const int tx = threadIdx.x & 31, ty = threadIdx.x >> 5;   // 32 x 8
  #pragma unroll
  for (int i = 0; i < 32; i += 8)
    tile[ty + i][tx] = W[(size_t)(kb + ty + i) * N + nb + tx];
  __syncthreads();
  #pragma unroll
  for (int i = 0; i < 32; i += 8)
    Wt[(size_t)(nb + ty + i) * K + kb + tx] = (__bf16)tile[tx][ty + i];
}

// ---------------------------------------------------------------------------
// Launcher
// ---------------------------------------------------------------------------
extern "C" void kernel_launch(void* const* d_in, const int* in_sizes, int n_in,
                              void* d_out, int out_size, void* d_ws, size_t ws_size,
                              hipStream_t stream) {
  (void)in_sizes; (void)n_in; (void)out_size; (void)ws_size;
  const float* x     = (const float*)d_in[0];
  const float* prev  = (const float*)d_in[1];
  const float* kqv_w = (const float*)d_in[2];
  const float* kqv_b = (const float*)d_in[3];
  const float* out_w = (const float*)d_in[4];
  const float* out_b = (const float*)d_in[5];
  const float* fc1_w = (const float*)d_in[6];
  const float* fc1_b = (const float*)d_in[7];
  const float* fc2_w = (const float*)d_in[8];
  const float* fc2_b = (const float*)d_in[9];
  const float* ln1_g = (const float*)d_in[10];
  const float* ln1_b = (const float*)d_in[11];
  const float* ln2_g = (const float*)d_in[12];
  const float* ln2_b = (const float*)d_in[13];

  // outputs: out [B,S,D] f32, then energy [B,H,S,S] f32
  float* outp   = (float*)d_out;
  float* energy = outp + (size_t)MROWS * DMODEL;

  // workspace carve-up (256B aligned slices)
  char* wp = (char*)d_ws;
  auto carve = [&](size_t bytes) {
    char* p = wp; wp += (bytes + 255) & ~(size_t)255; return (void*)p;
  };
  __bf16* kqvb   = (__bf16*)carve((size_t)MROWS * 3 * DMODEL * 2);  // bf16 kqv
  __bf16* ctxb   = (__bf16*)carve((size_t)MROWS * DMODEL * 2);      // bf16 ctx
  float*  pre1   = (float*) carve((size_t)MROWS * DMODEL * 4);
  float*  hbuf   = (float*) carve((size_t)MROWS * DMODEL * 4);
  __bf16* hbufb  = (__bf16*)carve((size_t)MROWS * DMODEL * 2);
  __bf16* fc1o   = (__bf16*)carve((size_t)MROWS * HIDDEN * 2);
  float*  pre2   = (float*) carve((size_t)MROWS * DMODEL * 4);
  __bf16* xb     = (__bf16*)carve((size_t)MROWS * DMODEL * 2);
  __bf16* kqvwt  = (__bf16*)carve((size_t)3 * DMODEL * DMODEL * 2); // [1536,512]
  __bf16* outwt  = (__bf16*)carve((size_t)DMODEL * DMODEL * 2);     // [512,512]
  __bf16* fc1wt  = (__bf16*)carve((size_t)HIDDEN * DMODEL * 2);     // [2048,512]
  __bf16* fc2wt  = (__bf16*)carve((size_t)DMODEL * HIDDEN * 2);     // [512,2048]

  const dim3 blk(256);

  // 0) one-shot bf16 conversions (weights transposed to [N,K])
  cvt_bf16_kernel<<<(MROWS * DMODEL / 4 + 255) / 256, blk, 0, stream>>>(
      x, xb, MROWS * DMODEL / 4);
  transpose_cvt_kernel<<<dim3(3 * DMODEL / 32, DMODEL / 32), blk, 0, stream>>>(
      kqv_w, kqvwt, DMODEL, 3 * DMODEL);
  transpose_cvt_kernel<<<dim3(DMODEL / 32, DMODEL / 32), blk, 0, stream>>>(
      out_w, outwt, DMODEL, DMODEL);
  transpose_cvt_kernel<<<dim3(HIDDEN / 32, DMODEL / 32), blk, 0, stream>>>(
      fc1_w, fc1wt, DMODEL, HIDDEN);
  transpose_cvt_kernel<<<dim3(DMODEL / 32, HIDDEN / 32), blk, 0, stream>>>(
      fc2_w, fc2wt, HIDDEN, DMODEL);

  // 1) kqv = x @ kqv_w + kqv_b                         (bf16 out)
  gemm_bf16_wmma<<<dim3(3 * DMODEL / BN, MROWS / BM), blk, 0, stream>>>(
      xb, kqvwt, kqv_b, nullptr, kqvb, MROWS, 3 * DMODEL, DMODEL, 0, 1);

  // 2) fused attention: energy out + softmax + ctx     (bf16 ctx)
  attn_flash<<<BATCH * NHEAD * (SEQ / 16), 32, 0, stream>>>(kqvb, prev, energy, ctxb);

  // 3) pre1 = ctx @ out_w + out_b + x                  (f32 out)
  gemm_bf16_wmma<<<dim3(DMODEL / BN, MROWS / BM), blk, 0, stream>>>(
      ctxb, outwt, out_b, x, pre1, MROWS, DMODEL, DMODEL, 0, 0);

  // 4) h = LN1(pre1)                                   (f32 + bf16 out)
  ln_kernel<<<MROWS, blk, 0, stream>>>(pre1, ln1_g, ln1_b, hbuf, hbufb, DMODEL);

  // 5) fc1o = gelu(h @ fc1_w + fc1_b)                  (bf16 out)
  gemm_bf16_wmma<<<dim3(HIDDEN / BN, MROWS / BM), blk, 0, stream>>>(
      hbufb, fc1wt, fc1_b, nullptr, fc1o, MROWS, HIDDEN, DMODEL, 1, 1);

  // 6) pre2 = fc1o @ fc2_w + fc2_b + h                 (f32 out)
  gemm_bf16_wmma<<<dim3(DMODEL / BN, MROWS / BM), blk, 0, stream>>>(
      fc1o, fc2wt, fc2_b, hbuf, pre2, MROWS, DMODEL, HIDDEN, 0, 0);

  // 7) out = LN2(pre2)
  ln_kernel<<<MROWS, blk, 0, stream>>>(pre2, ln2_g, ln2_b, outp, nullptr, DMODEL);
}
